// YAADLayer_78778290143542
// MI455X (gfx1250) — matrix operand order repaired and checked by reading, not compiled, not run on hardware
//
#include <hip/hip_runtime.h>
#include <hip/hip_bf16.h>
#include <math.h>

// ---------------------------------------------------------------------------
// Problem constants: B=8 S=4096 DM=1024 DH=128 LR=64 DI=256 CS=64 KW=4.
// q / qW / qn_g / qn_b are DEAD CODE in the reference (scan consumes only
// k,v,eta,delta,alpha; output = y@out_w.T * sigmoid(x@gate_w.T)).
// ---------------------------------------------------------------------------
#define BATCH 8
#define SEQ   4096
#define DM    1024
#define DH    128
#define LRNK  64
#define DI    256
#define CSZ   64
#define KWIN  4
#define NCH   (SEQ / CSZ)          // 64 chunks
#define MROWS (BATCH * SEQ)        // 32768 rows
#define NPACK 1472                 // 2*DH + 3*LRNK + DM
#define NPROJ 448                  // columns kept in ws (gate cols -> d_out)
#define KB32  32                   // WMMA bf16 K-step

typedef __bf16 bf16_t;
typedef __bf16 v16bf  __attribute__((ext_vector_type(16)));
typedef __bf16 bf16x8 __attribute__((ext_vector_type(8)));
typedef float  v8f    __attribute__((ext_vector_type(8)));
typedef int    v4i_   __attribute__((vector_size(16)));   // builtin's pointee type

// CDNA5 async global->LDS copies (ASYNCcnt-tracked); fall back to 16B vector
// copies if this toolchain doesn't declare the builtins.
#if defined(__HIP_DEVICE_COMPILE__) &&                                         \
    __has_builtin(__builtin_amdgcn_global_load_async_to_lds_b128) &&           \
    __has_builtin(__builtin_amdgcn_s_wait_asynccnt)
#define ASYNC_LDS 1
#else
#define ASYNC_LDS 0
#endif

// copy 16 bytes (8 bf16) global -> LDS
__device__ __forceinline__ void cp16(const bf16_t* g, bf16_t* l) {
#if ASYNC_LDS
  __builtin_amdgcn_global_load_async_to_lds_b128((v4i_*)g, (v4i_*)l, 0, 0);
#else
  *(bf16x8*)l = *(const bf16x8*)g;
#endif
}
__device__ __forceinline__ void cp_wait() {
#if ASYNC_LDS
  __builtin_amdgcn_s_wait_asynccnt(0);
#endif
}

__device__ __forceinline__ v8f v8f_zero() {
  v8f z = {0.f, 0.f, 0.f, 0.f, 0.f, 0.f, 0.f, 0.f};
  return z;
}

// D = A(16x32 bf16) * B(32x16 bf16) + C(16x16 f32)
__device__ __forceinline__ v8f wmma_bf16(v16bf a, v16bf b, v8f c) {
  return __builtin_amdgcn_wmma_f32_16x16x32_bf16(false, a, false, b, (short)0, c,
                                                 false, false);
}

// A fragment: rows row0..row0+15, K = k0..k0+31 from LDS row-major (stride lda).
// Lanes 0-15 hold K 0-15; lanes 16-31 hold K 16-31 (contiguous, matches B).
__device__ __forceinline__ v16bf load_a_frag(const bf16_t* As, int lda, int row0,
                                             int k0, int lane) {
  const int l16 = lane & 15;
  const int hi  = lane >> 4;
  const bf16_t* p = As + (row0 + l16) * lda + k0 + hi * 16;
  v16bf a;
#pragma unroll
  for (int t = 0; t < 16; ++t) a[t] = p[t];
  return a;
}

// B fragment from an N-major LDS tile Bst[n][k] (ldb = K stride).
// Lane L: column n0+(L%16), K half (L/16)*16 .. +15 (contiguous -> b128 loads).
__device__ __forceinline__ v16bf load_b_frag(const bf16_t* Bst, int ldb, int n0,
                                             int k0, int lane) {
  const int n  = n0 + (lane & 15);
  const int kb = k0 + (lane >> 4) * 16;
  const bf16_t* p = Bst + n * ldb + kb;
  v16bf b;
#pragma unroll
  for (int t = 0; t < 16; ++t) b[t] = p[t];
  return b;
}

__device__ __forceinline__ float sigmoidf_(float x) { return 1.0f / (1.0f + expf(-x)); }
__device__ __forceinline__ float softplusf_(float x) { return (x > 20.0f) ? x : log1pf(expf(x)); }
__device__ __forceinline__ float gelu_exact(float x) {
  return 0.5f * x * (1.0f + erff(x * 0.70710678118654752f));
}

// ---------------------------------------------------------------------------
// K0: pack projection weights bf16, N-MAJOR: wpT[n][k], n in
//   [0,128)=kW [128,256)=vW [256,320)=eta_w1 [320,384)=delta_w1
//   [384,448)=alpha_w1 [448,1472)=gate_w ; plus owp[n][k] = bf16(out_w).
// N-major makes every GEMM tile copy contiguous along K (16B chunks).
// ---------------------------------------------------------------------------
__global__ __launch_bounds__(256) void pack_weights_kernel(
    const float* __restrict__ kW, const float* __restrict__ vW,
    const float* __restrict__ ew1, const float* __restrict__ dw1,
    const float* __restrict__ aw1, const float* __restrict__ gw,
    const float* __restrict__ out_w,
    bf16_t* __restrict__ wpT, bf16_t* __restrict__ owp) {
  int idx = blockIdx.x * 256 + threadIdx.x;
  if (idx < DM * NPACK) {
    int n = idx / DM, k = idx % DM;
    float v;
    if      (n < 128) v = kW[n * DM + k];
    else if (n < 256) v = vW[(n - 128) * DM + k];
    else if (n < 320) v = ew1[(n - 256) * DM + k];
    else if (n < 384) v = dw1[(n - 320) * DM + k];
    else if (n < 448) v = aw1[(n - 384) * DM + k];
    else              v = gw[(long)(n - 448) * DM + k];
    wpT[idx] = (bf16_t)v;
  } else {
    int j = idx - DM * NPACK;
    if (j < DM * DH) owp[j] = (bf16_t)out_w[j];   // out_w already N-major (DM x DH)
  }
}

// K1: convert x (f32) -> bf16
__global__ __launch_bounds__(256) void cvt_x_kernel(const float* __restrict__ x,
                                                    bf16_t* __restrict__ xb) {
  long i = ((long)blockIdx.x * 256 + threadIdx.x) * 4;
  if (i >= (long)MROWS * DM) return;
  float4 f = *(const float4*)(x + i);
  xb[i + 0] = (bf16_t)f.x;
  xb[i + 1] = (bf16_t)f.y;
  xb[i + 2] = (bf16_t)f.z;
  xb[i + 3] = (bf16_t)f.w;
}

// ---------------------------------------------------------------------------
// K2: fused projection GEMM, C[32768,1472] = X[32768,1024] * W[1024,1472].
// BM=128, BN=64, 8 waves. Double-buffered ASYNC global->LDS staging pipeline:
//   wait(prev slab) -> barrier -> issue(next slab) -> WMMA(current slab).
// Columns < NPROJ -> proj scratch; columns >= NPROJ -> gate_raw (= d_out).
// ---------------------------------------------------------------------------
__global__ __launch_bounds__(256) void proj_gemm_kernel(
    const bf16_t* __restrict__ xb, const bf16_t* __restrict__ wpT,
    float* __restrict__ proj, float* __restrict__ gate_raw) {
  __shared__ __align__(16) bf16_t As[2][128][KB32];   // 16 KB
  __shared__ __align__(16) bf16_t Bs[2][64][KB32];    //  8 KB
  const int tid  = threadIdx.x;
  const int lane = tid & 31;
  const int wave = tid >> 5;
  const long m0  = (long)blockIdx.x * 128;
  const int  n0  = blockIdx.y * 64;

  auto stage = [&](int buf, int k0) {
    // A tile: 128 rows x 4 chunks of 8 bf16
#pragma unroll
    for (int t = tid; t < 512; t += 256) {
      int r = t >> 2, q = t & 3;
      cp16(&xb[(m0 + r) * DM + k0 + q * 8], &As[buf][r][q * 8]);
    }
    // B tile: 64 rows x 4 chunks (wpT is N-major, contiguous in k)
    {
      int r = tid >> 2, q = tid & 3;
      cp16(&wpT[(long)(n0 + r) * DM + k0 + q * 8], &Bs[buf][r][q * 8]);
    }
  };

  v8f acc[4];
#pragma unroll
  for (int j = 0; j < 4; ++j) acc[j] = v8f_zero();

  stage(0, 0);
  for (int k0 = 0; k0 < DM; k0 += KB32) {
    const int buf = (k0 >> 5) & 1;
    cp_wait();
    __syncthreads();
    if (k0 + KB32 < DM) stage(buf ^ 1, k0 + KB32);

    v16bf a = load_a_frag(&As[buf][0][0], KB32, wave * 16, 0, lane);
#pragma unroll
    for (int j = 0; j < 4; ++j) {
      v16bf b = load_b_frag(&Bs[buf][0][0], KB32, j * 16, 0, lane);
      acc[j] = wmma_bf16(a, b, acc[j]);
    }
    __syncthreads();   // frag reads done before next slab lands in buf^1
  }

  const int l16 = lane & 15, hi = lane >> 4;
#pragma unroll
  for (int j = 0; j < 4; ++j) {
    int n = n0 + j * 16 + l16;
#pragma unroll
    for (int r = 0; r < 8; ++r) {
      long m = m0 + wave * 16 + r + hi * 8;
      float val = acc[j][r];
      if (n < NPROJ) proj[m * NPROJ + n] = val;              // k/v/lr lanes
      else           gate_raw[m * DM + (n - NPROJ)] = val;   // gate pre-sigmoid
    }
  }
}

// ---------------------------------------------------------------------------
// K3: causal depthwise conv (KW=4) on k_raw/v_raw, LayerNorm(k), LR->DH
// projections + softplus/sigmoid.  One block per token, 128 threads.
// k is written as bf16 (feeds scan WMMA A directly).
// ---------------------------------------------------------------------------
__global__ __launch_bounds__(128) void conv_act_kernel(
    const float* __restrict__ proj,
    const float* __restrict__ conv_w, const float* __restrict__ conv_b,
    const float* __restrict__ kn_g, const float* __restrict__ kn_b,
    const float* __restrict__ eta_w2, const float* __restrict__ delta_w2,
    const float* __restrict__ alpha_w2,
    bf16_t* __restrict__ kout, float* __restrict__ vout,
    float* __restrict__ eta, float* __restrict__ delta, float* __restrict__ alpha) {
  const long m = blockIdx.x;
  const int h = threadIdx.x;
  const int s = (int)(m & (SEQ - 1));
  __shared__ float lr0[LRNK], lr1[LRNK], lr2[LRNK];
  __shared__ float red[DH];

  if (h < LRNK) {
    lr0[h] = proj[m * NPROJ + 256 + h];
    lr1[h] = proj[m * NPROJ + 320 + h];
    lr2[h] = proj[m * NPROJ + 384 + h];
  }

  float kacc = conv_b[h], vacc = conv_b[h];
#pragma unroll
  for (int j = 0; j < KWIN; ++j) {
    int ss = s - (KWIN - 1) + j;
    if (ss >= 0) {
      long mm = m - (KWIN - 1) + j;
      float w = conv_w[h * KWIN + j];
      kacc += w * proj[mm * NPROJ + h];
      vacc += w * proj[mm * NPROJ + 128 + h];
    }
  }
  __syncthreads();

  red[h] = kacc;
  __syncthreads();
  for (int off = 64; off > 0; off >>= 1) {
    if (h < off) red[h] += red[h + off];
    __syncthreads();
  }
  float mean = red[0] * (1.0f / DH);
  __syncthreads();
  float dv = kacc - mean;
  red[h] = dv * dv;
  __syncthreads();
  for (int off = 64; off > 0; off >>= 1) {
    if (h < off) red[h] += red[h + off];
    __syncthreads();
  }
  float var = red[0] * (1.0f / DH);
  __syncthreads();
  float kn = dv * rsqrtf(var + 1e-5f) * kn_g[h] + kn_b[h];

  float ea = 0.f, da = 0.f, aa = 0.f;
#pragma unroll
  for (int l = 0; l < LRNK; ++l) {
    ea += eta_w2[h * LRNK + l]   * lr0[l];
    da += delta_w2[h * LRNK + l] * lr1[l];
    aa += alpha_w2[h * LRNK + l] * lr2[l];
  }

  kout[m * DH + h]  = (bf16_t)kn;
  vout[m * DH + h]  = vacc;
  eta[m * DH + h]   = softplusf_(ea);
  delta[m * DH + h] = softplusf_(da);
  alpha[m * DH + h] = sigmoidf_(aa);
}

// ---------------------------------------------------------------------------
// K4: sequential fast-weight scan.  One block per batch (512 thr / 16 waves).
// State w1 (256x128), w2 (128x256) + transposed w2 copy all resident in LDS
// (~293 KB — needs CDNA5's 320 KB WGP LDS).  All chunk GEMMs are bf16 WMMA
// with contiguous b128 fragments; k chunks staged with async-to-LDS.
// ---------------------------------------------------------------------------
__global__ __launch_bounds__(512) void scan_kernel(
    const bf16_t* __restrict__ kbf, const float* __restrict__ varr,
    const float* __restrict__ eta, const float* __restrict__ delta,
    const float* __restrict__ alpha,
    const float* __restrict__ mw1, const float* __restrict__ mb1,
    const float* __restrict__ mw2, const float* __restrict__ mb2,
    bf16_t* __restrict__ ybf) {
  __shared__ __align__(16) bf16_t w1s[DI][DH];    // 64 KB  B of GEMM1 (N-major in i)
  __shared__ __align__(16) bf16_t w2s[DH][DI];    // 64 KB  B of GEMM2 (N-major in d)
  __shared__ __align__(16) bf16_t w2ts[DI][DH];   // 64 KB  B of GEMM3 (N-major in i)
  __shared__ __align__(16) bf16_t kcs[CSZ][DH];   // 16 KB
  __shared__ __align__(16) bf16_t acts[CSZ][DI];  // 32 KB
  __shared__ __align__(16) float  errf[CSZ][DH];  // 32 KB
  __shared__ __align__(16) bf16_t gps[CSZ][DH];   // 16 KB
  __shared__ float b1v[DI], b2v[DH], amv[DI], usv[DH], b1acc[DI];
  __shared__ float en2[CSZ];
  __shared__ float scal[2];                       // {alpha_w, eta_w}

  const int tid  = threadIdx.x;
  const int lane = tid & 31;
  const int wave = tid >> 5;
  const int l16  = lane & 15, hi = lane >> 4;
  const int crow = wave & 3;                      // row tile (64 rows = 4 tiles)
  const int wq   = wave >> 2;                     // column-tile phase
  const long base = (long)blockIdx.x * SEQ;

  for (int t = tid; t < DI * DH; t += 512) {
    int i = t / DH, d = t % DH;
    w1s[i][d] = (bf16_t)mw1[t];
  }
  for (int t = tid; t < DH * DI; t += 512) {
    int d = t / DI, i = t % DI;
    bf16_t w = (bf16_t)mw2[t];
    w2s[d][i] = w;
    w2ts[i][d] = w;
  }
  for (int t = tid; t < DI; t += 512) b1v[t] = mb1[t];
  if (tid < DH) b2v[tid] = mb2[tid];

  for (int ch = 0; ch < NCH; ++ch) {
    const long mb = base + (long)ch * CSZ;
    const long ml = mb + CSZ - 1;

    for (int t = tid; t < DI; t += 512) { amv[t] = 0.f; b1acc[t] = 0.f; }
    if (tid < DH) usv[tid] = 0.f;
    if (tid < CSZ) en2[tid] = 0.f;
    if (tid < 2) scal[tid] = 0.f;
    // stage k chunk (64x128 bf16) with 16B async copies
#pragma unroll
    for (int t = tid; t < CSZ * (DH / 8); t += 512) {
      int c = t >> 4, q = t & 15;
      cp16(&kbf[(mb + c) * DH + q * 8], &kcs[c][q * 8]);
    }
    cp_wait();
    __syncthreads();

    if (tid < DH) {
      atomicAdd(&scal[0], alpha[ml * DH + tid] * (1.0f / DH));
      atomicAdd(&scal[1], eta[ml * DH + tid] * (1.0f / DH));
    }

    // --- GEMM1: inter[c,i] = sum_d kc[c,d]*w1[i,d]; act = gelu(inter + b1)
    {
      v8f acc[4];
#pragma unroll
      for (int j = 0; j < 4; ++j) acc[j] = v8f_zero();
      for (int k0 = 0; k0 < DH; k0 += KB32) {
        v16bf a = load_a_frag(&kcs[0][0], DH, crow * 16, k0, lane);
#pragma unroll
        for (int j = 0; j < 4; ++j) {
          v16bf b = load_b_frag(&w1s[0][0], DH, (wq + j * 4) * 16, k0, lane);
          acc[j] = wmma_bf16(a, b, acc[j]);
        }
      }
#pragma unroll
      for (int j = 0; j < 4; ++j) {
        int i0 = (wq + j * 4) * 16 + l16;
#pragma unroll
        for (int r = 0; r < 8; ++r) {
          int c = crow * 16 + r + hi * 8;
          float g = gelu_exact(acc[j][r] + b1v[i0]);
          acts[c][i0] = (bf16_t)g;
          atomicAdd(&amv[i0], g * (1.0f / CSZ));
        }
      }
    }
    __syncthreads();

    // --- GEMM2: out[c,d] = sum_i act[c,i]*w2[d,i] + b2; y, err, |err|^2
    {
      v8f acc[2];
#pragma unroll
      for (int j = 0; j < 2; ++j) acc[j] = v8f_zero();
      for (int k0 = 0; k0 < DI; k0 += KB32) {
        v16bf a = load_a_frag(&acts[0][0], DI, crow * 16, k0, lane);
#pragma unroll
        for (int j = 0; j < 2; ++j) {
          v16bf b = load_b_frag(&w2s[0][0], DI, (wq + j * 4) * 16, k0, lane);
          acc[j] = wmma_bf16(a, b, acc[j]);
        }
      }
#pragma unroll
      for (int j = 0; j < 2; ++j) {
        int d0 = (wq + j * 4) * 16 + l16;
#pragma unroll
        for (int r = 0; r < 8; ++r) {
          int c = crow * 16 + r + hi * 8;
          float o = acc[j][r] + b2v[d0];
          long mrow = mb + c;
          ybf[mrow * DH + d0] = (bf16_t)o;
          float e = o - varr[mrow * DH + d0];
          errf[c][d0] = e;
          atomicAdd(&en2[c], e * e);
        }
      }
    }
    __syncthreads();

    // --- grad_pred = norm-clip(err); us = mean_c(grad_pred)
    for (int t = tid; t < CSZ * DH; t += 512) {
      int c = t >> 7, d = t & (DH - 1);
      float e = errf[c][d];
      float nrm = sqrtf(en2[c]);
      float dc = delta[(mb + c) * DH + d];
      float g = (nrm > dc) ? dc * e / (nrm + 1e-9f) : e;
      gps[c][d] = (bf16_t)g;
      atomicAdd(&usv[d], g * (1.0f / CSZ));
    }
    __syncthreads();

    // --- GEMM3: b1us[c,i] = sum_d gp[c,d]*w2[d,i]; column-mean -> b1acc
    {
      v8f acc[4];
#pragma unroll
      for (int j = 0; j < 4; ++j) acc[j] = v8f_zero();
      for (int k0 = 0; k0 < DH; k0 += KB32) {
        v16bf a = load_a_frag(&gps[0][0], DH, crow * 16, k0, lane);
#pragma unroll
        for (int j = 0; j < 4; ++j) {
          v16bf b = load_b_frag(&w2ts[0][0], DH, (wq + j * 4) * 16, k0, lane);
          acc[j] = wmma_bf16(a, b, acc[j]);
        }
      }
#pragma unroll
      for (int j = 0; j < 4; ++j) {
        int i0 = (wq + j * 4) * 16 + l16;
#pragma unroll
        for (int r = 0; r < 8; ++r)
          atomicAdd(&b1acc[i0], acc[j][r] * (1.0f / CSZ));
      }
    }
    __syncthreads();

    // --- rank-1 state updates (keep w2s and w2ts coherent)
    float aw = scal[0], ew = scal[1];
    for (int t = tid; t < DI * DH; t += 512) {
      int i = t / DH, d = t % DH;
      float w = (float)w1s[i][d];
      w1s[i][d] = (bf16_t)(aw * w - ew * amv[i] * usv[d]);
    }
    for (int t = tid; t < DH * DI; t += 512) {
      int d = t / DI, i = t % DI;
      float w = (float)w2s[d][i];
      bf16_t nv = (bf16_t)(aw * w - ew * usv[d] * amv[i]);
      w2s[d][i] = nv;
      w2ts[i][d] = nv;
    }
    if (tid < DH) {
      float ab = alpha[ml * DH + tid], eb = eta[ml * DH + tid];
      b2v[tid] = ab * b2v[tid] - eb * usv[tid];
    }
    for (int t = tid; t < DI; t += 512) b1v[t] = aw * b1v[t] - ew * b1acc[t];
    __syncthreads();
  }
}

// ---------------------------------------------------------------------------
// K5: out[m,n] = (y @ out_w.T)[m,n] * sigmoid(gate_raw[m,n])  (RMW on d_out)
// BM=128, BN=128, K=DH=128; both tiles staged with async-to-LDS.
// ---------------------------------------------------------------------------
__global__ __launch_bounds__(256) void out_gate_kernel(
    const bf16_t* __restrict__ ybf, const bf16_t* __restrict__ owp,
    float* __restrict__ out) {
  __shared__ __align__(16) bf16_t Ay[128][DH];  // 32 KB
  __shared__ __align__(16) bf16_t Bw[128][DH];  // 32 KB (owp is N-major)
  const int tid = threadIdx.x, lane = tid & 31, wave = tid >> 5;
  const long m0 = (long)blockIdx.x * 128;
  const int  n0 = blockIdx.y * 128;

#pragma unroll
  for (int t = tid; t < 128 * (DH / 8); t += 256) {
    int r = t >> 4, q = t & 15;
    cp16(&ybf[(m0 + r) * DH + q * 8], &Ay[r][q * 8]);
    cp16(&owp[(long)(n0 + r) * DH + q * 8], &Bw[r][q * 8]);
  }
  cp_wait();
  __syncthreads();

  v8f acc[8];
#pragma unroll
  for (int j = 0; j < 8; ++j) acc[j] = v8f_zero();
  for (int k0 = 0; k0 < DH; k0 += KB32) {
    v16bf a = load_a_frag(&Ay[0][0], DH, wave * 16, k0, lane);
#pragma unroll
    for (int j = 0; j < 8; ++j) {
      v16bf b = load_b_frag(&Bw[0][0], DH, j * 16, k0, lane);
      acc[j] = wmma_bf16(a, b, acc[j]);
    }
  }

  const int l16 = lane & 15, hi = lane >> 4;
#pragma unroll
  for (int j = 0; j < 8; ++j) {
    int n = n0 + j * 16 + l16;
#pragma unroll
    for (int r = 0; r < 8; ++r) {
      long m = m0 + wave * 16 + r + hi * 8;
      float gr = out[m * DM + n];           // gate pre-sigmoid written by K2
      out[m * DM + n] = acc[j][r] * sigmoidf_(gr);
    }
  }
}

// ---------------------------------------------------------------------------
extern "C" void kernel_launch(void* const* d_in, const int* in_sizes, int n_in,
                              void* d_out, int out_size, void* d_ws, size_t ws_size,
                              hipStream_t stream) {
  (void)in_sizes; (void)n_in; (void)out_size; (void)ws_size;
  const float* x        = (const float*)d_in[0];
  /* d_in[1] = qW : dead */
  const float* kW       = (const float*)d_in[2];
  const float* vW       = (const float*)d_in[3];
  const float* conv_w   = (const float*)d_in[4];
  const float* conv_b   = (const float*)d_in[5];
  const float* eta_w1   = (const float*)d_in[6];
  const float* eta_w2   = (const float*)d_in[7];
  const float* delta_w1 = (const float*)d_in[8];
  const float* delta_w2 = (const float*)d_in[9];
  const float* alpha_w1 = (const float*)d_in[10];
  const float* alpha_w2 = (const float*)d_in[11];
  const float* out_w    = (const float*)d_in[12];
  const float* gate_w   = (const float*)d_in[13];
  /* d_in[14]=qn_g, d_in[15]=qn_b : dead */
  const float* kn_g     = (const float*)d_in[16];
  const float* kn_b     = (const float*)d_in[17];
  const float* mw1      = (const float*)d_in[18];
  const float* mb1      = (const float*)d_in[19];
  const float* mw2      = (const float*)d_in[20];
  const float* mb2      = (const float*)d_in[21];
  /* d_in[22] = chunk_size (compile-time CSZ) */

  char* ws = (char*)d_ws;
  size_t off = 0;
  bf16_t* xb   = (bf16_t*)(ws + off); off += (size_t)MROWS * DM * sizeof(bf16_t);
  bf16_t* wpT  = (bf16_t*)(ws + off); off += (size_t)DM * NPACK * sizeof(bf16_t);
  bf16_t* owp  = (bf16_t*)(ws + off); off += (size_t)DM * DH * sizeof(bf16_t);
  float*  proj = (float*)(ws + off);  off += (size_t)MROWS * NPROJ * sizeof(float);
  bf16_t* kbuf = (bf16_t*)(ws + off); off += (size_t)MROWS * DH * sizeof(bf16_t);
  float*  vbuf = (float*)(ws + off);  off += (size_t)MROWS * DH * sizeof(float);
  float*  ebuf = (float*)(ws + off);  off += (size_t)MROWS * DH * sizeof(float);
  float*  dbuf = (float*)(ws + off);  off += (size_t)MROWS * DH * sizeof(float);
  float*  abuf = (float*)(ws + off);  off += (size_t)MROWS * DH * sizeof(float);
  bf16_t* ybuf = (bf16_t*)(ws + off); off += (size_t)MROWS * DH * sizeof(bf16_t);
  float*  outp = (float*)d_out;       // holds gate pre-sigmoid between K2 and K5

  pack_weights_kernel<<<(DM * (NPACK + DH) + 255) / 256, 256, 0, stream>>>(
      kW, vW, eta_w1, delta_w1, alpha_w1, gate_w, out_w, wpT, owp);

  cvt_x_kernel<<<(int)(((long)MROWS * DM / 4 + 255) / 256), 256, 0, stream>>>(x, xb);

  dim3 g2(MROWS / 128, NPACK / 64);
  proj_gemm_kernel<<<g2, 256, 0, stream>>>(xb, wpT, proj, outp);

  conv_act_kernel<<<MROWS, DH, 0, stream>>>(proj, conv_w, conv_b, kn_g, kn_b,
                                            eta_w2, delta_w2, alpha_w2,
                                            kbuf, vbuf, ebuf, dbuf, abuf);

  scan_kernel<<<BATCH, 512, 0, stream>>>(kbuf, vbuf, ebuf, dbuf, abuf,
                                         mw1, mb1, mw2, mb2, ybuf);

  dim3 g5(MROWS / 128, DM / 128);
  out_gate_kernel<<<g5, 256, 0, stream>>>(ybuf, owp, outp);
}
